// WindowAttention3D_20143396618483
// MI455X (gfx1250) — matrix operand order repaired and checked by reading, not compiled, not run on hardware
//
#include <hip/hip_runtime.h>

// ---------------------------------------------------------------------------
// Fused 3D shifted-window attention for MI455X (gfx1250, wave32, WMMA).
//
// One block = one 4x4x4 window (64 tokens). Everything (input tile, bf16
// weights, Q/K/Vt, scores, probabilities) lives in LDS (~194 KB of the
// 320 KB/WGP). All GEMMs use v_wmma_f32_16x16x32_bf16 with f32 accumulate.
// roll(-2)/roll(+2) are folded into the global load/store addressing.
// Weight tiles are copied global->LDS with GLOBAL_LOAD_ASYNC_TO_LDS_B128
// (ASYNCcnt) when the builtin is available, overlapped with the x-tile
// load+convert.
// ---------------------------------------------------------------------------

typedef __attribute__((ext_vector_type(16))) __bf16 v16bf;
typedef __attribute__((ext_vector_type(8)))  __bf16 v8bf;
typedef __attribute__((ext_vector_type(8)))  float  v8f;
typedef int vint4 __attribute__((vector_size(16)));   // matches builtin param

#define NW      16          // windows per axis (64/4)
#define DISP    2           // cyclic shift
#define SSTRIDE 66          // padded f32 row stride for scores (bank conflicts)

#define AS1 __attribute__((address_space(1)))
#define AS3 __attribute__((address_space(3)))

#if defined(__has_builtin)
#if __has_builtin(__builtin_amdgcn_global_load_async_to_lds_b128) && \
    __has_builtin(__builtin_amdgcn_s_wait_asynccnt)
#define USE_ASYNC_LDS 1
#endif
#endif

// ---- LDS layout (bytes) ---------------------------------------------------
#define OFF_WQKV 0                         // 288*96*2  = 55296
#define OFF_WOUT 55296                     //  96*96*2  = 18432
#define OFF_Q    73728                     //  64*96*2  = 12288  (row major [t][o])
#define OFF_K    86016                     //  64*96*2  = 12288  (row major [t][o])
#define OFF_VT   98304                     //  96*64*2  = 12288  (transposed [o][t])
#define OFF_XA   110592                    //  64*96*2  = 12288  (x tile, reused as attn-out)
#define OFF_P    122880                    // 3*64*64*2 = 24576  (probs, row major per head)
#define OFF_S    147456                    // 3*64*66*4 = 50688  (f32 scores, padded)
#define SMEM_BYTES 198144

// A fragment: 16x32 bf16, M rows striped per lane, K split per §7.12.2.
// lane(0..15): row=lane, K = {ksel*8 .. +7} and {16+ksel*8 .. +7}, ksel = lane>>4
__device__ __forceinline__ v16bf load_a(const __bf16* base, int stride,
                                        int m0, int k0, int lane) {
  const int row  = m0 + (lane & 15);
  const int ksel = (lane >> 4) & 1;
  const __bf16* p = base + row * stride + k0 + ksel * 8;
  v8bf lo = *(const v8bf*)(p);
  v8bf hi = *(const v8bf*)(p + 16);
  v16bf a;
#pragma unroll
  for (int i = 0; i < 8; ++i) { a[i] = lo[i]; a[8 + i] = hi[i]; }
  return a;
}

// B fragment: 32x16 bf16. Lane holds one column (N = n0 + lane&15) with
// K = {ksel*8 .. +7} and {16+ksel*8 .. +7}.  `base` is stored [n][k].
__device__ __forceinline__ v16bf load_b(const __bf16* base, int stride,
                                        int n0, int k0, int lane) {
  const int col  = n0 + (lane & 15);
  const int ksel = (lane >> 4) & 1;
  const __bf16* p = base + col * stride + k0 + ksel * 8;
  v8bf lo = *(const v8bf*)(p);
  v8bf hi = *(const v8bf*)(p + 16);
  v16bf b;
#pragma unroll
  for (int i = 0; i < 8; ++i) { b[i] = lo[i]; b[8 + i] = hi[i]; }
  return b;
}

__device__ __forceinline__ v8f wmma_bf16(v16bf a, v16bf b, v8f c) {
  return __builtin_amdgcn_wmma_f32_16x16x32_bf16(false, a, false, b,
                                                 (short)0, c, false, false);
}

// ---------------------------------------------------------------------------
// Prelude: fp32 weights -> bf16 in workspace (stays L2-resident afterwards).
// dst[0 .. 27647] = w_qkv (288x96), dst[27648 .. 36863] = w_out (96x96)
// ---------------------------------------------------------------------------
__global__ void cvt_weights(const float* __restrict__ wqkv,
                            const float* __restrict__ wout,
                            unsigned short* __restrict__ dst) {
  int i = blockIdx.x * 256 + threadIdx.x;
  float f;
  if (i < 27648)       f = wqkv[i];
  else if (i < 36864)  f = wout[i - 27648];
  else return;
  __bf16 h = (__bf16)f;
  dst[i] = __builtin_bit_cast(unsigned short, h);
}

// ---------------------------------------------------------------------------
// Main fused kernel: grid = 4096 windows, block = 256 threads (8 waves).
// ---------------------------------------------------------------------------
__global__ __launch_bounds__(256, 1)
void swin3d_fused(const float* __restrict__ x,
                  const float* __restrict__ b_out,
                  const float* __restrict__ xmask,
                  const float* __restrict__ ymask,
                  const float* __restrict__ zmask,
                  const unsigned short* __restrict__ wqkv_bf, // [288][96]
                  const unsigned short* __restrict__ wout_bf, // [96][96]
                  float* __restrict__ out) {
  extern __shared__ char smem[];
  __bf16* sWqkv = (__bf16*)(smem + OFF_WQKV);
  __bf16* sWout = (__bf16*)(smem + OFF_WOUT);
  __bf16* sQ    = (__bf16*)(smem + OFF_Q);
  __bf16* sK    = (__bf16*)(smem + OFF_K);
  __bf16* sVt   = (__bf16*)(smem + OFF_VT);
  __bf16* sXA   = (__bf16*)(smem + OFF_XA);   // x tile, later attention output
  __bf16* sP    = (__bf16*)(smem + OFF_P);
  float*  sS    = (float*) (smem + OFF_S);

  const int tid  = threadIdx.x;
  const int lane = tid & 31;
  const int wave = tid >> 5;
  const int blk  = blockIdx.x;
  const int wz = blk & 15, wy = (blk >> 4) & 15, wx = blk >> 8;

  // ---- Phase 0: weights -> LDS (async path if available), then the rolled
  //      x tile (fp32 -> bf16 via VALU). The async copy overlaps the latter.
#ifdef USE_ASYNC_LDS
  {
    const vint4* srcq = (const vint4*)wqkv_bf;     // 3456 x 16B
    const vint4* srco = (const vint4*)wout_bf;     // 1152 x 16B
    AS3 vint4* dstq = (AS3 vint4*)sWqkv;
    AS3 vint4* dsto = (AS3 vint4*)sWout;
    for (int i = tid; i < 3456; i += 256)
      __builtin_amdgcn_global_load_async_to_lds_b128(
          (AS1 vint4*)(srcq + i), dstq + i, 0, 0);
    for (int i = tid; i < 1152; i += 256)
      __builtin_amdgcn_global_load_async_to_lds_b128(
          (AS1 vint4*)(srco + i), dsto + i, 0, 0);
  }
#else
  {
    const uint4* src = (const uint4*)wqkv_bf;      // 3456 x 16B
    uint4*       dst = (uint4*)sWqkv;
    for (int i = tid; i < 3456; i += 256) dst[i] = src[i];
    const uint4* src2 = (const uint4*)wout_bf;     // 1152 x 16B
    uint4*       dst2 = (uint4*)sWout;
    for (int i = tid; i < 1152; i += 256) dst2[i] = src2[i];
  }
#endif

  for (int i = tid; i < 64 * 24; i += 256) {       // 1536 float4 loads
    const int t  = i / 24;
    const int c4 = (i % 24) << 2;
    const int lx = t >> 4, ly = (t >> 2) & 3, lz = t & 3;
    const int gx = (wx * 4 + lx + DISP) & 63;
    const int gy = (wy * 4 + ly + DISP) & 63;
    const int gz = (wz * 4 + lz + DISP) & 63;
    const float4 v = *(const float4*)(x + ((size_t)((gx * 64 + gy) * 64 + gz) * 96 + c4));
    __bf16* d = sXA + t * 96 + c4;
    d[0] = (__bf16)v.x; d[1] = (__bf16)v.y; d[2] = (__bf16)v.z; d[3] = (__bf16)v.w;
  }
#ifdef USE_ASYNC_LDS
  __builtin_amdgcn_s_wait_asynccnt(0);
#endif
  __syncthreads();

  // ---- Phase 1: QKV = X(64x96) @ Wqkv^T(96x288).
  // Each wave owns one M tile (A fragments hoisted into VGPRs) and marches
  // over 9 of the 18 O tiles. Q pre-scaled by 1/sqrt(head_dim); V stored
  // transposed ([o][t]).
  const float qscale = 0.17677669529663687f;       // 32^-0.5
  {
    const int m0 = (wave & 3) * 16;
    const v16bf a0 = load_a(sXA, 96, m0, 0,  lane);
    const v16bf a1 = load_a(sXA, 96, m0, 32, lane);
    const v16bf a2 = load_a(sXA, 96, m0, 64, lane);
    const int col = lane & 15;
    const int mb  = (lane >> 4) << 3;
    for (int nt = (wave >> 2); nt < 18; nt += 2) {
      const int o0 = nt * 16;
      v8f acc = {};
      acc = wmma_bf16(a0, load_b(sWqkv, 96, o0, 0,  lane), acc);
      acc = wmma_bf16(a1, load_b(sWqkv, 96, o0, 32, lane), acc);
      acc = wmma_bf16(a2, load_b(sWqkv, 96, o0, 64, lane), acc);
      if (o0 < 96) {                                // Q (scaled)
#pragma unroll
        for (int i = 0; i < 8; ++i)
          sQ[(m0 + mb + i) * 96 + o0 + col] = (__bf16)(acc[i] * qscale);
      } else if (o0 < 192) {                        // K
        const int oo = o0 - 96;
#pragma unroll
        for (int i = 0; i < 8; ++i)
          sK[(m0 + mb + i) * 96 + oo + col] = (__bf16)acc[i];
      } else {                                      // V -> transposed store
        const int oo = o0 - 192;
        v8bf tmp;
#pragma unroll
        for (int i = 0; i < 8; ++i) tmp[i] = (__bf16)acc[i];
        *(v8bf*)(sVt + (oo + col) * 64 + m0 + mb) = tmp;
      }
    }
  }
  __syncthreads();

  // ---- Phase 2: S = Q @ K^T per head (3 x (64x64), K-step = 32).
  // Wave owns one M tile; A fragment cached across the (h, nt) walk and
  // reloaded only when the head changes.
  {
    const int m0  = (wave & 3) * 16;
    const int col = lane & 15;
    const int mb  = (lane >> 4) << 3;
    int curh = -1;
    v16bf a;
    for (int j = (wave >> 2); j < 12; j += 2) {     // j = h*4 + nt
      const int h  = j >> 2;
      const int n0 = (j & 3) * 16;
      if (h != curh) { a = load_a(sQ, 96, m0, h * 32, lane); curh = h; }
      v16bf b = load_b(sK, 96, n0, h * 32, lane);
      v8f acc = {};
      acc = wmma_bf16(a, b, acc);
      float* S = sS + h * 64 * SSTRIDE;
#pragma unroll
      for (int i = 0; i < 8; ++i)
        S[(m0 + mb + i) * SSTRIDE + n0 + col] = acc[i];
    }
  }
  __syncthreads();

  // ---- Phase 3: masked softmax. 192 rows (3 heads x 64), one thread/row.
  const bool ax = (wx == NW - 1);
  const bool ay = (wy == NW - 1);
  const bool az = (wz == NW - 1);
  if (tid < 192) {
    const int h = tid / 64;
    const int i = tid % 64;
    float* S = sS + h * 64 * SSTRIDE + i * SSTRIDE;
    float mx = -__builtin_inff();
#pragma unroll 4
    for (int j = 0; j < 64; ++j) {
      float v = S[j];
      if (ax) v += xmask[i * 64 + j];
      if (ay) v += ymask[i * 64 + j];
      if (az) v += zmask[i * 64 + j];
      S[j] = v;
      mx = fmaxf(mx, v);
    }
    float sum = 0.f;
#pragma unroll 4
    for (int j = 0; j < 64; ++j) {
      float e = __expf(S[j] - mx);
      S[j] = e;
      sum += e;
    }
    const float inv = 1.0f / sum;
    __bf16* P = sP + h * 4096 + i * 64;
#pragma unroll 4
    for (int j = 0; j < 64; ++j) P[j] = (__bf16)(S[j] * inv);
  }
  __syncthreads();

  // ---- Phase 4: O = P(64x64) @ V(64x32) per head. 24 tiles, K-steps = 2.
  // Result overwrites sXA (x tile is dead) as row-major [t][h*32+d].
  for (int tt = wave; tt < 24; tt += 8) {
    const int h  = tt >> 3;
    const int r  = tt & 7;
    const int m0 = (r & 3) * 16;
    const int n0 = (r >> 2) * 16;
    v8f acc = {};
#pragma unroll
    for (int kk = 0; kk < 2; ++kk) {
      v16bf a = load_a(sP + h * 4096, 64, m0, kk * 32, lane);
      v16bf b = load_b(sVt + h * 32 * 64, 64, n0, kk * 32, lane);
      acc = wmma_bf16(a, b, acc);
    }
    const int col = lane & 15;
    const int mb  = (lane >> 4) << 3;
#pragma unroll
    for (int i = 0; i < 8; ++i)
      sXA[(m0 + mb + i) * 96 + h * 32 + n0 + col] = (__bf16)acc[i];
  }
  __syncthreads();

  // ---- Phase 5: out = O(64x96) @ Wout^T(96x96) + bias, stored with roll(+2).
  // Wave owns one M tile; A fragments hoisted.
  {
    const int m0 = (wave & 3) * 16;
    const v16bf a0 = load_a(sXA, 96, m0, 0,  lane);
    const v16bf a1 = load_a(sXA, 96, m0, 32, lane);
    const v16bf a2 = load_a(sXA, 96, m0, 64, lane);
    const int col = lane & 15;
    const int mb  = (lane >> 4) << 3;
    for (int nt = (wave >> 2); nt < 6; nt += 2) {
      const int c0 = nt * 16;
      v8f acc = {};
      acc = wmma_bf16(a0, load_b(sWout, 96, c0, 0,  lane), acc);
      acc = wmma_bf16(a1, load_b(sWout, 96, c0, 32, lane), acc);
      acc = wmma_bf16(a2, load_b(sWout, 96, c0, 64, lane), acc);
      const int c = c0 + col;
      const float bias = b_out[c];
#pragma unroll
      for (int i = 0; i < 8; ++i) {
        const int t  = m0 + mb + i;
        const int lx = t >> 4, ly = (t >> 2) & 3, lz = t & 3;
        const int gx = (wx * 4 + lx + DISP) & 63;
        const int gy = (wy * 4 + ly + DISP) & 63;
        const int gz = (wz * 4 + lz + DISP) & 63;
        out[(size_t)((gx * 64 + gy) * 64 + gz) * 96 + c] = acc[i] + bias;
      }
    }
  }
}

// ---------------------------------------------------------------------------
// Host launcher. Inputs (setup_inputs order):
//   0:x(f32 64^3x96) 1:w_qkv(288x96) 2:w_out(96x96) 3:b_out(96)
//   4:x_mask 5:y_mask 6:z_mask (all f32 64x64)
// ---------------------------------------------------------------------------
extern "C" void kernel_launch(void* const* d_in, const int* in_sizes, int n_in,
                              void* d_out, int out_size, void* d_ws, size_t ws_size,
                              hipStream_t stream) {
  (void)in_sizes; (void)n_in; (void)out_size; (void)ws_size;
  const float* x     = (const float*)d_in[0];
  const float* wqkv  = (const float*)d_in[1];
  const float* wout  = (const float*)d_in[2];
  const float* b_out = (const float*)d_in[3];
  const float* xm    = (const float*)d_in[4];
  const float* ym    = (const float*)d_in[5];
  const float* zm    = (const float*)d_in[6];
  float*       outp  = (float*)d_out;

  unsigned short* wq_bf = (unsigned short*)d_ws;        // 27648 halves
  unsigned short* wo_bf = wq_bf + 288 * 96;             //  9216 halves

  cvt_weights<<<144, 256, 0, stream>>>(wqkv, wout, wq_bf);
  swin3d_fused<<<4096, 256, SMEM_BYTES, stream>>>(x, b_out, xm, ym, zm,
                                                  wq_bf, wo_bf, outp);
}
